// RecommendationModel_64725157151217
// MI455X (gfx1250) — compile-verified, matrix operand order checked by reading, not run on hardware
//
#include <hip/hip_runtime.h>
#include <hip/hip_bf16.h>

#define VOCAB 100000
#define DIM   128
#define BATCH 2048
#define HIST  50
#define TOPK  10
#define NT    5          // N-tiles per wave in the GEMM
#define WAVES_PER_BLK 8  // 256 threads, wave32

typedef __bf16 v16bf __attribute__((ext_vector_type(16)));
typedef float  v8f   __attribute__((ext_vector_type(8)));

// ---------------------------------------------------------------------------
// Kernel 1: convert label embedding table f32 -> bf16 (25.6 MB, L2-resident)
// ---------------------------------------------------------------------------
__global__ void cvt_label_bf16(const float* __restrict__ in,
                               __bf16* __restrict__ out, int npairs) {
    int i = blockIdx.x * blockDim.x + threadIdx.x;
    if (i < npairs) {
        float2 f = ((const float2*)in)[i];
        out[2 * i + 0] = (__bf16)f.x;
        out[2 * i + 1] = (__bf16)f.y;
    }
}

// ---------------------------------------------------------------------------
// Kernel 2: context encoder: embedding-bag mean + dense projection -> bf16
// one block (128 threads) per batch row
// ---------------------------------------------------------------------------
__global__ void __launch_bounds__(DIM) ctx_encode(
    const int* __restrict__ ids, const float* __restrict__ etab,
    const float* __restrict__ W, const float* __restrict__ bias,
    __bf16* __restrict__ ctxbf) {
    __shared__ float pooled[DIM];
    const int b = blockIdx.x;
    const int d = threadIdx.x;
    const int* row = ids + b * HIST;
    float s = 0.0f;
    for (int h = 0; h < HIST; ++h)                  // row[h] is wave-uniform
        s += etab[(long)row[h] * DIM + d];          // coalesced 512B gathers
    pooled[d] = s * (1.0f / (float)HIST);
    __syncthreads();
    float acc = bias[d];
    #pragma unroll 8
    for (int k = 0; k < DIM; ++k)
        acc += pooled[k] * W[k * DIM + d];
    ctxbf[b * DIM + d] = (__bf16)acc;
}

// ---------------------------------------------------------------------------
// Kernel 3: logits GEMM, v_wmma_f32_16x16x32_bf16.
// One wave owns one 16-row M tile and NT consecutive 16-col N tiles.
// A (16x128) held in 32 VGPRs, reused across N tiles. B operand = label rows
// (column-major B == row-major label^T), contiguous 32B loads per lane.
// ---------------------------------------------------------------------------
__global__ void __launch_bounds__(256, 4) logits_gemm(
    const __bf16* __restrict__ A,    // [BATCH, DIM] ctx bf16
    const __bf16* __restrict__ Bm,   // [VOCAB, DIM] label bf16
    float* __restrict__ out) {       // [BATCH, VOCAB]
    const int lane   = threadIdx.x & 31;
    const int wave   = threadIdx.x >> 5;
    const int half   = lane >> 4;    // 0: K lo 16, 1: K hi 16 of each chunk
    const int r16    = lane & 15;
    const int m0     = blockIdx.y * 16;
    const int ntile0 = (blockIdx.x * WAVES_PER_BLK + wave) * NT;

    // A tile: lane r16 holds row m0+r16, halves split 16 contiguous K each.
    const __bf16* ap = A + (m0 + r16) * DIM + half * 16;
    v16bf a0 = *(const v16bf*)(ap + 0);
    v16bf a1 = *(const v16bf*)(ap + 32);
    v16bf a2 = *(const v16bf*)(ap + 64);
    v16bf a3 = *(const v16bf*)(ap + 96);

    for (int t = 0; t < NT; ++t) {
        const int n0 = (ntile0 + t) * 16;
        if (n0 >= VOCAB) break;      // wave-uniform: EXEC stays all-1s
        const __bf16* bp = Bm + (long)(n0 + r16) * DIM + half * 16;
        v16bf b0 = *(const v16bf*)(bp + 0);
        v16bf b1 = *(const v16bf*)(bp + 32);
        v16bf b2 = *(const v16bf*)(bp + 64);
        v16bf b3 = *(const v16bf*)(bp + 96);
        v8f c = {};
        c = __builtin_amdgcn_wmma_f32_16x16x32_bf16(false, a0, false, b0, (short)0, c, false, false);
        c = __builtin_amdgcn_wmma_f32_16x16x32_bf16(false, a1, false, b1, (short)0, c, false, false);
        c = __builtin_amdgcn_wmma_f32_16x16x32_bf16(false, a2, false, b2, (short)0, c, false, false);
        c = __builtin_amdgcn_wmma_f32_16x16x32_bf16(false, a3, false, b3, (short)0, c, false, false);
        // D layout: VGPR r, lanes 0-15 -> row m0+r col n0+lane;
        //           lanes 16-31 -> row m0+r+8. Two 64B segments per store.
        float* op = out + (long)(m0 + half * 8) * VOCAB + (n0 + r16);
        #pragma unroll
        for (int r = 0; r < 8; ++r)
            op[(long)r * VOCAB] = c[r];
    }
}

// ---------------------------------------------------------------------------
// Kernel 4: top-k (k=10) per row. One 256-thread block per batch row.
// Phase 1: per-thread sorted top-10 over a strided slice.
// Phase 2: log2(256) rounds of sorted-list merges in LDS.
// ---------------------------------------------------------------------------
__global__ void __launch_bounds__(256) topk_rows(
    const float* __restrict__ logits,
    float* __restrict__ out_ids, float* __restrict__ out_scores) {
    __shared__ float sv[256 * TOPK];
    __shared__ int   si[256 * TOPK];
    const int b   = blockIdx.x;
    const int tid = threadIdx.x;
    const float* row = logits + (long)b * VOCAB;

    float vals[TOPK];
    int   idxs[TOPK];
    #pragma unroll
    for (int k = 0; k < TOPK; ++k) { vals[k] = -3.4e38f; idxs[k] = -1; }

    // ascending order: vals[0] is the current minimum of the kept set
    for (int i = tid; i < VOCAB; i += 256) {
        float v = row[i];
        if (v > vals[0]) {
            int p = 0;
            while (p < TOPK - 1 && vals[p + 1] < v) {
                vals[p] = vals[p + 1]; idxs[p] = idxs[p + 1]; ++p;
            }
            vals[p] = v; idxs[p] = i;
        }
    }
    #pragma unroll
    for (int k = 0; k < TOPK; ++k) {
        sv[tid * TOPK + k] = vals[k];
        si[tid * TOPK + k] = idxs[k];
    }
    __syncthreads();

    for (int stride = 128; stride >= 1; stride >>= 1) {
        if (tid < stride) {
            float mv[TOPK]; int mi[TOPK];
            float* av  = sv + tid * TOPK;            int* ai  = si + tid * TOPK;
            float* bv2 = sv + (tid + stride) * TOPK; int* bi2 = si + (tid + stride) * TOPK;
            int i = TOPK - 1, j = TOPK - 1;
            for (int k = TOPK - 1; k >= 0; --k) {
                bool takeB;
                if (i < 0)      takeB = true;
                else if (j < 0) takeB = false;
                else            takeB = (bv2[j] > av[i]);
                if (takeB) { mv[k] = bv2[j]; mi[k] = bi2[j]; --j; }
                else       { mv[k] = av[i];  mi[k] = ai[i];  --i; }
            }
            for (int k = 0; k < TOPK; ++k) { av[k] = mv[k]; ai[k] = mi[k]; }
        }
        __syncthreads();
    }

    if (tid == 0) {
        for (int k = 0; k < TOPK; ++k) {
            out_ids[b * TOPK + k]    = (float)si[TOPK - 1 - k];  // descending
            out_scores[b * TOPK + k] = sv[TOPK - 1 - k];
        }
    }
}

// ---------------------------------------------------------------------------
extern "C" void kernel_launch(void* const* d_in, const int* in_sizes, int n_in,
                              void* d_out, int out_size, void* d_ws, size_t ws_size,
                              hipStream_t stream) {
    const int*   ids  = (const int*)d_in[0];    // [2048, 50]
    const float* ctab = (const float*)d_in[1];  // [100000, 128]
    const float* ltab = (const float*)d_in[2];  // [100000, 128]
    const float* W    = (const float*)d_in[3];  // [128, 128]
    const float* bias = (const float*)d_in[4];  // [128]
    // d_in[5] = top_k scalar (10), compiled in as TOPK

    float* logits     = (float*)d_out;                       // [2048, 100000]
    float* out_ids    = logits + (size_t)BATCH * VOCAB;      // [2048, 10]
    float* out_scores = out_ids + (size_t)BATCH * TOPK;      // [2048, 10]

    __bf16* labelbf = (__bf16*)d_ws;                                   // 25.6 MB
    __bf16* ctxbf   = (__bf16*)((char*)d_ws + (size_t)VOCAB * DIM * 2);// 0.5 MB

    // 1) label table f32 -> bf16 (fits in 192MB L2 for reuse across M tiles)
    {
        int npairs = VOCAB * DIM / 2;
        int blocks = (npairs + 255) / 256;
        cvt_label_bf16<<<blocks, 256, 0, stream>>>(ltab, labelbf, npairs);
    }
    // 2) context encoder
    ctx_encode<<<BATCH, DIM, 0, stream>>>(ids, ctab, W, bias, ctxbf);
    // 3) WMMA GEMM: 128 M-tiles x 6250 N-tiles
    {
        int ntiles   = VOCAB / 16;                        // 6250
        int per_blk  = WAVES_PER_BLK * NT;                // 40 tiles per block
        dim3 grid((ntiles + per_blk - 1) / per_blk, BATCH / 16);
        logits_gemm<<<grid, 256, 0, stream>>>(ctxbf, labelbf, logits);
    }
    // 4) top-k retrieval
    topk_rows<<<BATCH, 256, 0, stream>>>(logits, out_ids, out_scores);
}
